// ScaledDotProduct_71262097375552
// MI455X (gfx1250) — matrix operand order repaired and checked by reading, not compiled, not run on hardware
//
#include <hip/hip_runtime.h>
#include <hip/hip_fp16.h>

typedef __attribute__((ext_vector_type(16))) _Float16 v16h;
typedef __attribute__((ext_vector_type(8)))  _Float16 v8h;
typedef __attribute__((ext_vector_type(4)))  _Float16 v4h;
typedef __attribute__((ext_vector_type(2)))  _Float16 v2h;
typedef __attribute__((ext_vector_type(2)))  __fp16   v2fp;
typedef __attribute__((ext_vector_type(8)))  float    v8f;
typedef __attribute__((ext_vector_type(4)))  float    v4f;

constexpr int BATCH = 8;
constexpr int SEQ   = 2048;
constexpr int DIM   = 128;
constexpr int QT    = 64;       // q rows per block (4 waves x 16)
constexpr int KT    = 64;       // keys per staged tile
constexpr int SKP   = DIM + 8;  // sK row stride (f16), conflict-free b128
constexpr int VTP   = KT + 8;   // sVt row stride (f16)
constexpr int PP    = KT + 8;   // P scratch row stride (f16)

__device__ __forceinline__ v16h join16(const _Float16* p0, const _Float16* p1) {
    v8h a = *(const v8h*)p0;
    v8h b = *(const v8h*)p1;
    v16h r;
#pragma unroll
    for (int j = 0; j < 8; ++j) { r[j] = a[j]; r[8 + j] = b[j]; }
    return r;
}

__device__ __forceinline__ void stage_tile(const float* __restrict__ kt,
                                           const float* __restrict__ vt,
                                           _Float16* __restrict__ dK,
                                           _Float16* __restrict__ dVt,
                                           int tid) {
    // 64 rows x 128 cols f32 per matrix; 128 threads x 16 float4 each.
#pragma unroll
    for (int i = 0; i < 16; ++i) {
        int lin = tid + 128 * i;        // 0..2047
        int row = lin >> 5;             // 32 float4 per row
        int c4  = lin & 31;
        v4f kd = *(const v4f*)(kt + row * DIM + c4 * 4);
        v4f vd = *(const v4f*)(vt + row * DIM + c4 * 4);
        // packed f32->f16 (v_cvt_pk_rtz_f16_f32): 2 VALU per float4
        v2h k01 = __builtin_bit_cast(v2h, __builtin_amdgcn_cvt_pkrtz(kd[0], kd[1]));
        v2h k23 = __builtin_bit_cast(v2h, __builtin_amdgcn_cvt_pkrtz(kd[2], kd[3]));
        v4h kh;
        kh[0] = k01[0]; kh[1] = k01[1]; kh[2] = k23[0]; kh[3] = k23[1];
        *(v4h*)(dK + row * SKP + c4 * 4) = kh;      // K row-major
#pragma unroll
        for (int j = 0; j < 4; ++j)                 // V transposed: [d][key]
            dVt[(c4 * 4 + j) * VTP + row] = (_Float16)vd[j];
    }
}

__global__ __launch_bounds__(128)
void fa_fwd_kernel(const float* __restrict__ q, const float* __restrict__ k,
                   const float* __restrict__ v, const float* __restrict__ mask,
                   float* __restrict__ out) {
    __shared__ _Float16 sK [2][KT  * SKP];
    __shared__ _Float16 sVt[2][DIM * VTP];
    __shared__ _Float16 sP [4][16 * PP];

    const int tid  = threadIdx.x;
    const int wave = tid >> 5;
    const int lane = tid & 31;
    const int ln   = lane & 15;
    const int hi   = lane >> 4;

    const int b     = blockIdx.y;
    const int qbase = blockIdx.x * QT + wave * 16;

    const float scale = 0.08838834764831845f;  // 1/sqrt(128)

    // ---- Q tile -> 4 A-fragments (16x32 f16), scale folded in ----
    v16h qa[4];
    const float* qrow = q + ((size_t)b * SEQ + qbase + ln) * DIM;
#pragma unroll
    for (int c = 0; c < 4; ++c) {
        int d0 = 32 * c + hi * 8;
        int d1 = d0 + 16;
        v4f a0 = *(const v4f*)(qrow + d0);
        v4f a1 = *(const v4f*)(qrow + d0 + 4);
        v4f b0 = *(const v4f*)(qrow + d1);
        v4f b1 = *(const v4f*)(qrow + d1 + 4);
#pragma unroll
        for (int j = 0; j < 4; ++j) {
            qa[c][j]      = (_Float16)(a0[j] * scale);
            qa[c][4 + j]  = (_Float16)(a1[j] * scale);
            qa[c][8 + j]  = (_Float16)(b0[j] * scale);
            qa[c][12 + j] = (_Float16)(b1[j] * scale);
        }
    }

    // ---- accumulators / online-softmax state (rows j+8*hi) ----
    v8f o[8];
#pragma unroll
    for (int n = 0; n < 8; ++n)
#pragma unroll
        for (int j = 0; j < 8; ++j) o[n][j] = 0.0f;
    float m_[8], l_[8];
#pragma unroll
    for (int j = 0; j < 8; ++j) { m_[j] = -1.0e30f; l_[j] = 0.0f; }

    const float* kbase = k + (size_t)b * SEQ * DIM;
    const float* vbase = v + (size_t)b * SEQ * DIM;
    const float* mbase = mask + (size_t)b * SEQ;

    constexpr int NT = SEQ / KT;  // 32 key tiles
    stage_tile(kbase, vbase, sK[0], sVt[0], tid);
    __syncthreads();

    for (int t = 0; t < NT; ++t) {
        const int bb = t & 1;
        if (t + 1 < NT)  // prefetch next tile into the other buffer
            stage_tile(kbase + (size_t)(t + 1) * KT * DIM,
                       vbase + (size_t)(t + 1) * KT * DIM,
                       sK[bb ^ 1], sVt[bb ^ 1], tid);

        // mask values for this tile's 64 key columns (col = f*16 + ln)
        float mk[4];
#pragma unroll
        for (int f = 0; f < 4; ++f) mk[f] = mbase[t * KT + f * 16 + ln];

        // ---- GEMM1: S(16x64) = Q(16x128) @ K^T, 16 WMMAs, 1-deep pipeline ----
        const _Float16* kfp = &sK[bb][ln * SKP + hi * 8];
        v8f sf[4];
#pragma unroll
        for (int f = 0; f < 4; ++f) {
            v8f acc;
#pragma unroll
            for (int j = 0; j < 8; ++j) acc[j] = 0.0f;
            const _Float16* kp0 = kfp + (f * 16) * SKP;
            v16h kb_cur = join16(kp0, kp0 + 16);
#pragma unroll
            for (int c = 0; c < 4; ++c) {
                v16h kb_nxt;
                if (c < 3) {  // issue next fragment's loads before this WMMA
                    const _Float16* kp = kp0 + (c + 1) * 32;
                    kb_nxt = join16(kp, kp + 16);
                }
                acc = __builtin_amdgcn_wmma_f32_16x16x32_f16(
                    false, qa[c], false, kb_cur, (short)0, acc, false, false);
                if (c < 3) kb_cur = kb_nxt;
            }
#pragma unroll
            for (int j = 0; j < 8; ++j) acc[j] += mk[f];
            sf[f] = acc;
        }

        // ---- online softmax (row = j + 8*hi; reduce across 16-lane half) ----
        float mt[8];
#pragma unroll
        for (int j = 0; j < 8; ++j)
            mt[j] = fmaxf(fmaxf(sf[0][j], sf[1][j]), fmaxf(sf[2][j], sf[3][j]));
#pragma unroll
        for (int off = 1; off < 16; off <<= 1)
#pragma unroll
            for (int j = 0; j < 8; ++j) mt[j] = fmaxf(mt[j], __shfl_xor(mt[j], off));

        float al[8];
#pragma unroll
        for (int j = 0; j < 8; ++j) {
            float mn = fmaxf(m_[j], mt[j]);
            al[j] = __expf(m_[j] - mn);
            m_[j] = mn;
        }
        float rs[8];
#pragma unroll
        for (int j = 0; j < 8; ++j) rs[j] = 0.0f;
#pragma unroll
        for (int f = 0; f < 4; ++f)
#pragma unroll
            for (int j = 0; j < 8; ++j) {
                float p = __expf(sf[f][j] - m_[j]);
                sf[f][j] = p;
                rs[j] += p;
            }
#pragma unroll
        for (int off = 1; off < 16; off <<= 1)
#pragma unroll
            for (int j = 0; j < 8; ++j) rs[j] += __shfl_xor(rs[j], off);
#pragma unroll
        for (int j = 0; j < 8; ++j) l_[j] = l_[j] * al[j] + rs[j];
#pragma unroll
        for (int n = 0; n < 8; ++n)
#pragma unroll
            for (int j = 0; j < 8; ++j) o[n][j] *= al[j];

        // ---- P: C-layout -> A-layout via per-wave LDS round trip ----
#pragma unroll
        for (int f = 0; f < 4; ++f)
#pragma unroll
            for (int j = 0; j < 8; ++j)
                sP[wave][(j + hi * 8) * PP + f * 16 + ln] = (_Float16)sf[f][j];
        asm volatile("s_wait_dscnt 0" ::: "memory");  // per-wave LDS RAW

        v16h pa[2];
#pragma unroll
        for (int kc = 0; kc < 2; ++kc) {
            const _Float16* pb = &sP[wave][ln * PP + kc * 32 + hi * 8];
            pa[kc] = join16(pb, pb + 16);
        }

        // ---- GEMM2: O(16x128) += P(16x64) @ V, 16 WMMAs, 1-deep pipeline ----
        const _Float16* vfp = &sVt[bb][ln * VTP + hi * 8];
        {
            v16h vb_cur = join16(vfp, vfp + 16);  // (dt=0, kc=0)
#pragma unroll
            for (int i = 0; i < 16; ++i) {
                int dt = i >> 1, kc = i & 1;
                v16h vb_nxt;
                if (i < 15) {
                    int dt2 = (i + 1) >> 1, kc2 = (i + 1) & 1;
                    const _Float16* vp = vfp + (dt2 * 16) * VTP + kc2 * 32;
                    vb_nxt = join16(vp, vp + 16);
                }
                o[dt] = __builtin_amdgcn_wmma_f32_16x16x32_f16(
                    false, pa[kc], false, vb_cur, (short)0, o[dt], false, false);
                if (i < 15) vb_cur = vb_nxt;
            }
        }

        __syncthreads();  // staging(bb^1) done + compute(bb) done
    }

    // ---- epilogue: normalize by l; non-temporal stores (result never re-read) ----
    float* op = out + ((size_t)b * SEQ + qbase) * DIM;
#pragma unroll
    for (int j = 0; j < 8; ++j) {
        float inv = 1.0f / l_[j];
        int row = j + hi * 8;
#pragma unroll
        for (int n = 0; n < 8; ++n)
            __builtin_nontemporal_store(o[n][j] * inv,
                                        op + (size_t)row * DIM + n * 16 + ln);
    }
}

extern "C" void kernel_launch(void* const* d_in, const int* in_sizes, int n_in,
                              void* d_out, int out_size, void* d_ws, size_t ws_size,
                              hipStream_t stream) {
    (void)in_sizes; (void)n_in; (void)out_size; (void)d_ws; (void)ws_size;
    const float* q  = (const float*)d_in[0];
    const float* k  = (const float*)d_in[1];
    const float* v  = (const float*)d_in[2];
    const float* mk = (const float*)d_in[3];
    float* o = (float*)d_out;
    dim3 grid(SEQ / QT, BATCH);
    fa_fwd_kernel<<<grid, 128, 0, stream>>>(q, k, v, mk, o);
}
